// DecodeNFlowFunc_83571473645847
// MI455X (gfx1250) — matrix operand order repaired and checked by reading, not compiled, not run on hardware
//
#include <hip/hip_runtime.h>
#include <math.h>

// ---------------------------------------------------------------------------
// Fused RealNVP-style flow decode for MI455X (gfx1250), wave32 + WMMA f16.
// ---------------------------------------------------------------------------

typedef __attribute__((ext_vector_type(16))) _Float16 v16h;
typedef __attribute__((ext_vector_type(8)))  float    v8f;

#define DIMX   128
#define HDIM   32
#define DDIM   64
#define STW    127     // 2*(128-64)-1
#define SM1    63
#define NBLK   4
#define NSAMP  524288
#define NTILES (NSAMP / 16)   // 32768
#define WAVES  8
#define TPB    (WAVES * 32)

// workspace (d_ws) layout, bytes
#define WS_PERM_OFF 0                    // 4*128 ints  = 2048 B
#define WS_FRAG_OFF 2048                 // 124 frags * 1024 B = 126976 B
#define NFRAG       124
#define FRAG_HALFS  512                  // 32 lanes * 16 halfs
#define WS_BIAS_OFF (2048 + NFRAG * 1024)        // 129024
#define BIAS_FLOATS 1728
#define WS_TOTAL    (WS_BIAS_OFF + BIAS_FLOATS * 4)  // 135936

// per-wave LDS buffers (floats)
#define ST_STRIDE  132                   // 16 rows  (132%64==4 -> conflict free)
#define HB_STRIDE  68
#define STATE_F    (16 * ST_STRIDE)      // 2112
#define SBUF_F     (16 * ST_STRIDE)      // 2112
#define HBUF_F     (16 * HB_STRIDE)      // 1088
#define WAVE_F     (STATE_F + SBUF_F + HBUF_F)   // 5312 floats = 21248 B
#define SMEM_MAIN  (WS_TOTAL + WAVES * WAVE_F * 4)  // 305920 B  (<320KB WGP LDS)

// fast math helpers -------------------------------------------------------
#if __has_builtin(__builtin_amdgcn_tanhf)
__device__ __forceinline__ float fast_tanhf(float x) { return __builtin_amdgcn_tanhf(x); }
#else
__device__ __forceinline__ float fast_tanhf(float x) { return tanhf(x); }
#endif

__device__ __forceinline__ float reluf(float x) {
  return __builtin_amdgcn_fmed3f(x, 0.0f, __builtin_inff());  // v_med3_f32
}

__device__ __forceinline__ float softplusf(float x) {
  // max(x,0) + log(1 + exp(-|x|)), native v_exp/v_log
  return fmaxf(x, 0.0f) + __logf(1.0f + __expf(-fabsf(x)));
}

// ---------------------------------------------------------------------------
// Prep kernel 1: numpy-exact MT19937 permutations for seeds 0..NBLK-1.
// ---------------------------------------------------------------------------
__global__ void perm_kernel(int* __restrict__ perm_out) {
  __shared__ unsigned mt[NBLK][624];
  const int tid = threadIdx.x;
  if (tid >= NBLK) return;
  unsigned* m = mt[tid];
  // mt19937_seed (numpy): key[pos]=seed; seed = 1812433253*(seed^(seed>>30)) + pos + 1
  unsigned seed = (unsigned)tid;
  for (int i = 0; i < 624; ++i) {
    m[i] = seed;
    seed = 1812433253u * (seed ^ (seed >> 30)) + (unsigned)i + 1u;
  }
  int idx = 624;
  int arr[DIMX];
  for (int c = 0; c < DIMX; ++c) arr[c] = c;
  // Fisher-Yates with numpy random_interval (mask + rejection) semantics
  for (int i = DIMX - 1; i > 0; --i) {
    unsigned mask = (unsigned)i;
    mask |= mask >> 1; mask |= mask >> 2; mask |= mask >> 4;
    mask |= mask >> 8; mask |= mask >> 16;
    unsigned j;
    do {
      if (idx >= 624) {  // twist
        for (int t = 0; t < 624; ++t) {
          unsigned y = (m[t] & 0x80000000u) | (m[(t + 1) % 624] & 0x7fffffffu);
          unsigned v = m[(t + 397) % 624] ^ (y >> 1);
          if (y & 1u) v ^= 0x9908b0dfu;
          m[t] = v;
        }
        idx = 0;
      }
      unsigned y = m[idx++];
      y ^= y >> 11; y ^= (y << 7) & 0x9d2c5680u;
      y ^= (y << 15) & 0xefc60000u; y ^= y >> 18;
      j = y & mask;
    } while (j > (unsigned)i);
    int tv = arr[i]; arr[i] = arr[(int)j]; arr[(int)j] = tv;
  }
  for (int c = 0; c < DIMX; ++c) perm_out[tid * DIMX + c] = arr[c];
}

// ---------------------------------------------------------------------------
// Prep kernel 2: repack weights into WMMA B-fragment order (f16), pad biases.
// B layout (16-bit, 32x16, wave32): lane L, element e -> N = L&15,
//   K = (e&7) + 8*(L>>4) + 16*(e>>3).  One fragment = 512 halfs, lane-major (32B per lane).
// Fragment order: [0,1]=fw0(N-tiles) [2,3]=fw1 [4..11]=fw2,
//   then per coupling k: 4x cw0 (nt-major, ks inner), 2x cw1, 8x cw2.
// ---------------------------------------------------------------------------
__global__ void wprep_kernel(const float* __restrict__ fw0, const float* __restrict__ fw1,
                             const float* __restrict__ fw2, const float* __restrict__ cw0,
                             const float* __restrict__ cw1, const float* __restrict__ cw2,
                             const float* __restrict__ fb0, const float* __restrict__ fb1,
                             const float* __restrict__ fb2, const float* __restrict__ cb0,
                             const float* __restrict__ cb1, const float* __restrict__ cb2,
                             char* __restrict__ ws) {
  const int gid = blockIdx.x * blockDim.x + threadIdx.x;
  _Float16* frag = (_Float16*)(ws + WS_FRAG_OFF);
  float* bias = (float*)(ws + WS_BIAS_OFF);
  if (gid < NFRAG * FRAG_HALFS) {
    const int f = gid >> 9, i = gid & 511;
    const int lane = i >> 4, e = i & 15;
    const int nl = lane & 15;
    const int kk = (e & 7) + 8 * (lane >> 4) + 16 * (e >> 3);
    const float* W; int Kin, Nout, Koff = 0, Noff;
    if (f < 2)       { W = fw0; Kin = 2;  Nout = 32;  Noff = 16 * f; }
    else if (f < 4)  { W = fw1; Kin = 32; Nout = 32;  Noff = 16 * (f - 2); }
    else if (f < 12) { W = fw2; Kin = 32; Nout = 126; Noff = 16 * (f - 4); }
    else {
      const int f2 = f - 12, k = f2 / 14, r = f2 % 14;
      if (r < 4)      { W = cw0 + k * (32 * 64);  Kin = 64; Nout = 32;  Noff = 16 * (r >> 1); Koff = 32 * (r & 1); }
      else if (r < 6) { W = cw1 + k * (32 * 32);  Kin = 32; Nout = 32;  Noff = 16 * (r - 4); }
      else            { W = cw2 + k * (127 * 32); Kin = 32; Nout = 127; Noff = 16 * (r - 6); }
    }
    const int n = Noff + nl, ka = Koff + kk;
    float v = (n < Nout && ka < Kin) ? W[n * Kin + ka] : 0.0f;
    frag[gid] = (_Float16)v;
  } else if (gid < NFRAG * FRAG_HALFS + BIAS_FLOATS) {
    const int b = gid - NFRAG * FRAG_HALFS;
    float v = 0.0f;
    if (b < 32)        v = fb0[b];
    else if (b < 64)   v = fb1[b - 32];
    else if (b < 192)  { int c = b - 64; if (c < 126) v = fb2[c]; }
    else {
      const int b2 = b - 192, k = b2 / 192, r = b2 % 192;
      if (r < 32)      v = cb0[k * 32 + r];
      else if (r < 64) v = cb1[k * 32 + (r - 32)];
      else             { int c = r - 64; if (c < 127) v = cb2[k * 127 + c]; }
    }
    bias[b] = v;
  }
}

// ---------------------------------------------------------------------------
// WMMA helpers
// ---------------------------------------------------------------------------
__device__ __forceinline__ v8f wmma32(v16h a, v16h b, v8f c) {
  return __builtin_amdgcn_wmma_f32_16x16x32_f16(false, a, false, b, (short)0, c,
                                                false, false);
}

// A layout (16-bit, 16x32): lane L, element e -> M = L&15,
//   K = koff + (e&7) + 8*(L>>4) + 16*(e>>3)
__device__ __forceinline__ v16h loadA(const float* buf, int stride, int m, int g,
                                      int koff) {
  const float* p = buf + m * stride + koff + 8 * g;
  v16h a;
#pragma unroll
  for (int e = 0; e < 8; ++e) a[e] = (_Float16)p[e];
#pragma unroll
  for (int e = 0; e < 8; ++e) a[8 + e] = (_Float16)p[16 + e];
  return a;
}

__device__ __forceinline__ v8f biasAcc(float bv) {
  v8f c;
#pragma unroll
  for (int r = 0; r < 8; ++r) c[r] = bv;
  return c;
}

// C layout (f32, 16x16): lane L, element r -> M = r + 8*(L>>4), N = L&15
__device__ __forceinline__ void storeC(float* buf, int stride, int g, int nl,
                                       int noff, v8f acc, bool relu) {
#pragma unroll
  for (int r = 0; r < 8; ++r) {
    float v = acc[r];
    if (relu) v = reluf(v);
    buf[(r + 8 * g) * stride + noff + nl] = v;
  }
}

// ---------------------------------------------------------------------------
// Main fused kernel: 1 wave = 16 rows, full flow kept in LDS.
// ---------------------------------------------------------------------------
__global__ __launch_bounds__(TPB) void flow_kernel(const float* __restrict__ Z,
                                                   const char* __restrict__ ws,
                                                   float* __restrict__ out) {
  extern __shared__ char smem[];
  const int tid = threadIdx.x;
  const int lane = tid & 31, w = tid >> 5;
  const int g = lane >> 4, nl = lane & 15;

  // stage perms + weight fragments + biases to LDS (coalesced)
  {
    const float4* src = (const float4*)ws;
    float4* dst = (float4*)smem;
    for (int i = tid; i < WS_TOTAL / 16; i += TPB) dst[i] = src[i];
  }
  __syncthreads();

  const int* perms = (const int*)smem;
  const _Float16* frag = (const _Float16*)(smem + WS_FRAG_OFF);
  const float* bias = (const float*)(smem + WS_BIAS_OFF);
  float* wbuf = (float*)(smem + WS_TOTAL) + w * WAVE_F;
  float* state = wbuf;                    // 16 x 132
  float* sbuf = wbuf + STATE_F;           // 16 x 132 (st / MLP head scratch)
  float* hbuf = wbuf + STATE_F + SBUF_F;  // 16 x 68  (hA cols0..31, hB cols36..67)

  const int tile = blockIdx.x * WAVES + w;
  const int row0 = tile * 16;

  // ---------------- initial MLP: z(2) -> 32 -> 32 -> 126 -------------------
  const float z0 = Z[(row0 + nl) * 2 + 0];
  const float z1 = Z[(row0 + nl) * 2 + 1];
  v16h a0;
#pragma unroll
  for (int e = 0; e < 16; ++e) a0[e] = (_Float16)0.0f;
  if (g == 0) { a0[0] = (_Float16)z0; a0[1] = (_Float16)z1; }  // K=0,1 live only here
#pragma unroll
  for (int nt = 0; nt < 2; ++nt) {
    v8f acc = biasAcc(bias[0 + 16 * nt + nl]);
    v16h b = *(const v16h*)(frag + (0 + nt) * FRAG_HALFS + lane * 16);
    acc = wmma32(a0, b, acc);
    storeC(hbuf, HB_STRIDE, g, nl, 16 * nt, acc, true);
  }
  {
    v16h a = loadA(hbuf, HB_STRIDE, nl, g, 0);
#pragma unroll
    for (int nt = 0; nt < 2; ++nt) {
      v8f acc = biasAcc(bias[32 + 16 * nt + nl]);
      v16h b = *(const v16h*)(frag + (2 + nt) * FRAG_HALFS + lane * 16);
      acc = wmma32(a, b, acc);
      storeC(hbuf, HB_STRIDE, g, nl, 36 + 16 * nt, acc, true);
    }
  }
  {
    v16h a = loadA(hbuf, HB_STRIDE, nl, g, 36);
#pragma unroll
    for (int nt = 0; nt < 8; ++nt) {
      v8f acc = biasAcc(bias[64 + 16 * nt + nl]);
      v16h b = *(const v16h*)(frag + (4 + nt) * FRAG_HALFS + lane * 16);
      acc = wmma32(a, b, acc);
      storeC(sbuf, ST_STRIDE, g, nl, 16 * nt, acc, false);
    }
  }
  // state = concat([z, head])
  if (g == 0) { state[nl * ST_STRIDE + 0] = z0; state[nl * ST_STRIDE + 1] = z1; }
  for (int idx = lane; idx < 16 * 126; idx += 32) {
    int r = idx / 126, c = idx % 126;
    state[r * ST_STRIDE + 2 + c] = sbuf[r * ST_STRIDE + c];
  }

  // ---------------- 4 blocks x (perm + 2 couplings) ------------------------
  for (int ii = 0; ii < NBLK; ++ii) {
    const int* pm = perms + ii * DIMX;
    // in-place column permutation, chunked by disjoint row groups
    for (int chunk = 0; chunk < 4; ++chunk) {
      float v[16];
#pragma unroll
      for (int t = 0; t < 16; ++t) {
        int idx = chunk * 512 + t * 32 + lane;
        int r = idx >> 7, c = idx & 127;
        v[t] = state[r * ST_STRIDE + pm[c]];
      }
#pragma unroll
      for (int t = 0; t < 16; ++t) {
        int idx = chunk * 512 + t * 32 + lane;
        int r = idx >> 7, c = idx & 127;
        state[r * ST_STRIDE + c] = v[t];
      }
    }
    for (int jj = 0; jj < 2; ++jj) {
      const int k = 2 * ii + jj;
      const int fb = 12 + 14 * k;
      const int bb = 192 + 192 * k;
      // GEMM1: x1(16x64) @ cw0^T -> h(16x32), relu
      {
        v16h ak0 = loadA(state, ST_STRIDE, nl, g, 0);
        v16h ak1 = loadA(state, ST_STRIDE, nl, g, 32);
#pragma unroll
        for (int nt = 0; nt < 2; ++nt) {
          v8f acc = biasAcc(bias[bb + 16 * nt + nl]);
          v16h b0 = *(const v16h*)(frag + (fb + nt * 2 + 0) * FRAG_HALFS + lane * 16);
          acc = wmma32(ak0, b0, acc);
          v16h b1 = *(const v16h*)(frag + (fb + nt * 2 + 1) * FRAG_HALFS + lane * 16);
          acc = wmma32(ak1, b1, acc);
          storeC(hbuf, HB_STRIDE, g, nl, 16 * nt, acc, true);
        }
      }
      // GEMM2: 32 -> 32, relu
      {
        v16h a = loadA(hbuf, HB_STRIDE, nl, g, 0);
#pragma unroll
        for (int nt = 0; nt < 2; ++nt) {
          v8f acc = biasAcc(bias[bb + 32 + 16 * nt + nl]);
          v16h b = *(const v16h*)(frag + (fb + 4 + nt) * FRAG_HALFS + lane * 16);
          acc = wmma32(a, b, acc);
          storeC(hbuf, HB_STRIDE, g, nl, 36 + 16 * nt, acc, true);
        }
      }
      // GEMM3: 32 -> 127 (padded 128) -> sbuf
      {
        v16h a = loadA(hbuf, HB_STRIDE, nl, g, 36);
#pragma unroll
        for (int nt = 0; nt < 8; ++nt) {
          v8f acc = biasAcc(bias[bb + 64 + 16 * nt + nl]);
          v16h b = *(const v16h*)(frag + (fb + 6 + nt) * FRAG_HALFS + lane * 16);
          acc = wmma32(a, b, acc);
          storeC(sbuf, ST_STRIDE, g, nl, 16 * nt, acc, false);
        }
      }
      // s = 0.1*tanh(st[:, :63]) in-place  (hardware v_tanh_f32 when available)
      for (int idx = lane; idx < 16 * SM1; idx += 32) {
        int r = idx / SM1, c = idx % SM1;
        float* p = sbuf + r * ST_STRIDE + c;
        *p = 0.1f * fast_tanhf(*p);
      }
      // partial row sums of s, both half-waves active:
      //   g==0: cols 0..31 -> col 128 ; g==1: cols 32..62 -> col 129
      {
        float s = 0.0f;
        const int c0 = g ? 32 : 0, c1 = g ? SM1 : 32;
        for (int c = c0; c < c1; ++c) s += sbuf[nl * ST_STRIDE + c];
        sbuf[nl * ST_STRIDE + 128 + g] = s;
      }
      // trans = x2 * exp(s) + t  -> hbuf cols 0..63
      for (int idx = lane; idx < 16 * 64; idx += 32) {
        int r = idx >> 6, c = idx & 63;
        float s = (c < SM1) ? sbuf[r * ST_STRIDE + c]
                            : -(sbuf[r * ST_STRIDE + 128] + sbuf[r * ST_STRIDE + 129]);
        float t = sbuf[r * ST_STRIDE + SM1 + c];
        float x2v = state[r * ST_STRIDE + 64 + c];
        hbuf[r * HB_STRIDE + c] = x2v * __expf(s) + t;
      }
      // state = [trans, x1]   (copy x1 -> upper half first, row-chunked)
      for (int chunk = 0; chunk < 4; ++chunk) {
        float v[8];
#pragma unroll
        for (int t = 0; t < 8; ++t) {
          int idx = chunk * 256 + t * 32 + lane;
          int r = idx >> 6, c = idx & 63;
          v[t] = state[r * ST_STRIDE + c];
        }
#pragma unroll
        for (int t = 0; t < 8; ++t) {
          int idx = chunk * 256 + t * 32 + lane;
          int r = idx >> 6, c = idx & 63;
          state[r * ST_STRIDE + 64 + c] = v[t];
        }
      }
      for (int idx = lane; idx < 16 * 64; idx += 32) {
        int r = idx >> 6, c = idx & 63;
        state[r * ST_STRIDE + c] = hbuf[r * HB_STRIDE + c];
      }
    }
  }

  // ---------------- softplus + coalesced float4 store ----------------------
  float* outp = out + (size_t)row0 * DIMX;
#pragma unroll
  for (int t = 0; t < 16; ++t) {
    int idx4 = t * 32 + lane;            // 512 float4s over 16x128 tile
    int r = idx4 >> 5;
    int c = (idx4 & 31) * 4;
    float4 vv = *(const float4*)(state + r * ST_STRIDE + c);
    float4 o;
    o.x = softplusf(vv.x); o.y = softplusf(vv.y);
    o.z = softplusf(vv.z); o.w = softplusf(vv.w);
    *(float4*)(outp + (size_t)idx4 * 4) = o;
  }
}

// ---------------------------------------------------------------------------
extern "C" void kernel_launch(void* const* d_in, const int* in_sizes, int n_in,
                              void* d_out, int out_size, void* d_ws, size_t ws_size,
                              hipStream_t stream) {
  (void)in_sizes; (void)n_in; (void)out_size; (void)ws_size;
  const float* Z   = (const float*)d_in[0];
  const float* fw0 = (const float*)d_in[1];
  const float* fb0 = (const float*)d_in[2];
  const float* fw1 = (const float*)d_in[3];
  const float* fb1 = (const float*)d_in[4];
  const float* fw2 = (const float*)d_in[5];
  const float* fb2 = (const float*)d_in[6];
  const float* cw0 = (const float*)d_in[7];
  const float* cb0 = (const float*)d_in[8];
  const float* cw1 = (const float*)d_in[9];
  const float* cb1 = (const float*)d_in[10];
  const float* cw2 = (const float*)d_in[11];
  const float* cb2 = (const float*)d_in[12];
  char* ws = (char*)d_ws;

  perm_kernel<<<1, 32, 0, stream>>>((int*)(ws + WS_PERM_OFF));

  const int total = NFRAG * FRAG_HALFS + BIAS_FLOATS;
  wprep_kernel<<<(total + 255) / 256, 256, 0, stream>>>(
      fw0, fw1, fw2, cw0, cw1, cw2, fb0, fb1, fb2, cb0, cb1, cb2, ws);

  (void)hipFuncSetAttribute(reinterpret_cast<const void*>(flow_kernel),
                            hipFuncAttributeMaxDynamicSharedMemorySize,
                            (int)SMEM_MAIN);
  flow_kernel<<<NTILES / WAVES, TPB, SMEM_MAIN, stream>>>(Z, ws, (float*)d_out);
}
